// RelMultiHeadedAttention_171798692216
// MI455X (gfx1250) — matrix operand order repaired and checked by reading, not compiled, not run on hardware
//
#include <hip/hip_runtime.h>
#include <hip/hip_bf16.h>

// Problem constants (match reference)
#define HH     16
#define DHEAD  64
#define DMODEL 1024
#define BB     4
#define QL     1024
#define KL     2048
// base-2 softmax: fold log2(e) into the score scale
#define SCALE2 0.36067376022224085f   // 0.25 * log2(e)
#define NEG_F  -1.0e9f

typedef __attribute__((ext_vector_type(16))) _Float16 v16h;
typedef __attribute__((ext_vector_type(8)))  float    v8f;

// ---------------------------------------------------------------------------
// WMMA operand loaders, CDNA5 ISA 7.12.2 layouts (wave32).
// A (16x32 f16, MxK): lane L -> M = L%16, kbase = (L/16)*8;
//   elems 0..7 -> K = kbase+e, elems 8..15 -> K = 16+kbase+(e-8).
// B (32x16 f16, KxN): lane L -> N = L%16, same K order as A.
// C/D (16x16 f32): VGPR r, lane L -> M = r + 8*(L/16), N = L%16.
// ---------------------------------------------------------------------------
static __device__ inline v16h wmma_ld_a(const _Float16* base, int ld) {
  int lane = threadIdx.x & 31;
  int m  = lane & 15;
  int kb = (lane >> 4) * 8;
  const _Float16* p = base + (size_t)m * ld + kb;
  v16h a;
#pragma unroll
  for (int e = 0; e < 8; ++e) a[e] = p[e];
#pragma unroll
  for (int e = 0; e < 8; ++e) a[8 + e] = p[16 + e];
  return a;
}

// B operand from memory holding B^T: mem[n][k], contiguous in k.
static __device__ inline v16h wmma_ld_bT(const _Float16* base, int ld) {
  int lane = threadIdx.x & 31;
  int n  = lane & 15;
  int kb = (lane >> 4) * 8;
  const _Float16* p = base + (size_t)n * ld + kb;
  v16h b;
#pragma unroll
  for (int e = 0; e < 8; ++e) b[e] = p[e];
#pragma unroll
  for (int e = 0; e < 8; ++e) b[8 + e] = p[16 + e];
  return b;
}

static __device__ inline v8f wmma32(v16h a, v16h b, v8f c) {
  return __builtin_amdgcn_wmma_f32_16x16x32_f16(false, a, false, b,
                                                (short)0, c, false, false);
}

// ---------------------------------------------------------------------------
// fp32 -> fp16 conversion, 4 elements/thread (all sizes are multiples of 4)
// ---------------------------------------------------------------------------
__global__ void cvt_f16(const float* __restrict__ x, _Float16* __restrict__ y,
                        int n4) {
  int i = (blockIdx.x * blockDim.x + threadIdx.x) * 4;
  if (i < n4) {
    float4 v = *(const float4*)(x + i);
    y[i + 0] = (_Float16)v.x;
    y[i + 1] = (_Float16)v.y;
    y[i + 2] = (_Float16)v.z;
    y[i + 3] = (_Float16)v.w;
  }
}

// ---------------------------------------------------------------------------
// Projection GEMM: out[row, e] = sum_d X[row, d] * W[e, d]  (X @ W^T)
// One wave computes a 32x64 block: 2 A-tiles x 4 B-tiles, 8 WMMAs per K-step.
// A traffic /4 and W traffic /2 vs one 16x16 tile per wave.
// MODE 0: q-projection -> (q + r_w_bias) and q, head-major f16 [b,h,l,d]
// MODE 1: k-projection -> head-major f16 [b,h,j,d]
// MODE 2: output projection -> f32 [b,l,e] (final result)
// MODE 3: v-projection -> transposed head-major f16 [b,h,d,j]
// ---------------------------------------------------------------------------
template <int MODE>
__global__ __launch_bounds__(128) void gemm_wmma(
    const _Float16* __restrict__ X, const _Float16* __restrict__ W,
    int rowsTotal, int L,
    float* __restrict__ outF,
    _Float16* __restrict__ outA, _Float16* __restrict__ outB,
    const float* __restrict__ bias) {
  int wid  = threadIdx.x >> 5;
  int lane = threadIdx.x & 31;
  int tile = blockIdx.x * 4 + wid;
  const int ntn = DMODEL / 64;          // 16 column strips (one head each)
  int tm = tile / ntn;
  int tn = tile - tm * ntn;
  int row0 = tm * 32;
  if (row0 >= rowsTotal) return;
  int e0 = tn * 64;

  const _Float16* xr0 = X + (size_t)row0 * DMODEL;
  const _Float16* xr1 = xr0 + (size_t)16 * DMODEL;
  v8f c[2][4] = {};
  for (int k0 = 0; k0 < DMODEL; k0 += 32) {
    v16h a0 = wmma_ld_a(xr0 + k0, DMODEL);
    v16h a1 = wmma_ld_a(xr1 + k0, DMODEL);
#pragma unroll
    for (int t = 0; t < 4; ++t) {
      v16h bm = wmma_ld_bT(W + (size_t)(e0 + t * 16) * DMODEL + k0, DMODEL);
      c[0][t] = wmma32(a0, bm, c[0][t]);
      c[1][t] = wmma32(a1, bm, c[1][t]);
    }
  }

  int g = lane >> 4, n = lane & 15;
  int h = tn;                            // strip == head (64 == DHEAD)
#pragma unroll
  for (int u = 0; u < 2; ++u) {
#pragma unroll
    for (int t = 0; t < 4; ++t) {
      int d = t * 16 + n;
      float bv = (MODE == 0) ? bias[h * DHEAD + d] : 0.0f;
#pragma unroll
      for (int r = 0; r < 8; ++r) {
        int row = row0 + u * 16 + r + 8 * g;
        float v = c[u][t][r];
        if (MODE == 2) {
          outF[(size_t)row * DMODEL + e0 + t * 16 + n] = v;
        } else {
          int bidx = row / L;
          int l = row - bidx * L;
          if (MODE == 3) {
            outA[((size_t)(bidx * HH + h) * DHEAD + d) * KL + l] = (_Float16)v;
          } else {
            size_t o = ((size_t)(bidx * HH + h) * L + l) * DHEAD + d;
            if (MODE == 0) {
              outA[o] = (_Float16)(v + bv);
              outB[o] = (_Float16)v;
            } else {
              outA[o] = (_Float16)v;
            }
          }
        }
      }
    }
  }
}

// ---------------------------------------------------------------------------
// Flash-style relative attention. One wave handles one (b, h, 16-row q tile).
// shift() identity:  BD[i,j] = (c < KLEN) ? B_[i,c] + rb[c] : 0,  c = j+1023-i
// The 64-wide shifted-B window advances by exactly 32 per chunk, so it is kept
// as a ring buffer in LDS (column index c & 63) and only the two new 16-col
// tiles are computed per chunk.
// ---------------------------------------------------------------------------
__global__ __launch_bounds__(128) void attn_wmma(
    const _Float16* __restrict__ qw, const _Float16* __restrict__ qp,
    const _Float16* __restrict__ kh, const _Float16* __restrict__ vhT,
    const _Float16* __restrict__ re, const float* __restrict__ rbias,
    const unsigned char* __restrict__ mask, _Float16* __restrict__ av16) {
  __shared__ _Float16 Pt[4][16][40];   // P tile staging (C-layout -> A-layout)
  __shared__ float    Bb[4][16][68];   // shifted-B ring window (64 + pad)

  int wid  = threadIdx.x >> 5;
  int lane = threadIdx.x & 31;
  int g = lane >> 4, n = lane & 15;

  int tile = blockIdx.x * 4 + wid;
  int b   = tile / (HH * (QL / 16));
  int rem = tile - b * (HH * (QL / 16));
  int h   = rem / (QL / 16);
  int i0  = (rem - h * (QL / 16)) * 16;

  const _Float16* qwp = qw + ((size_t)(b * HH + h) * QL + i0) * DHEAD;
  const _Float16* qpp = qp + ((size_t)(b * HH + h) * QL + i0) * DHEAD;
  const _Float16* khp = kh + (size_t)(b * HH + h) * KL * DHEAD;
  const _Float16* vhp = vhT + (size_t)(b * HH + h) * DHEAD * KL;
  const _Float16* rep = re + (size_t)h * KL * DHEAD;
  const float* rbp = rbias + (size_t)h * KL;
  const unsigned char* mp = mask + (size_t)b * QL * KL;

  // Persistent A operands: (q + r_w_bias) and plain q, K-split d 0..31/32..63
  v16h aQW0 = wmma_ld_a(qwp, DHEAD);
  v16h aQW1 = wmma_ld_a(qwp + 32, DHEAD);
  v16h aQP0 = wmma_ld_a(qpp, DHEAD);
  v16h aQP1 = wmma_ld_a(qpp + 32, DHEAD);

  float m_r[8], l_r[8];
  v8f acc[4] = {};
#pragma unroll
  for (int r = 0; r < 8; ++r) { m_r[r] = -3.0e38f; l_r[r] = 0.0f; }

  // Compute one 16-col tile of (q . re^T + rb) into the ring window.
  auto btile = [&](int ct) {
    if (ct < KL) {
      v8f bc = {};
      bc = wmma32(aQP0, wmma_ld_bT(rep + (size_t)ct * DHEAD, DHEAD), bc);
      bc = wmma32(aQP1, wmma_ld_bT(rep + (size_t)ct * DHEAD + 32, DHEAD), bc);
      float rv = rbp[ct + n];
      int col = (ct & 63) + n;
#pragma unroll
      for (int r = 0; r < 8; ++r) Bb[wid][r + 8 * g][col] = bc[r] + rv;
    }
  };

  // Prime the lower half of the first window [cw, cw+32).
  int cw = (QL - 16) - i0;             // 16-aligned, >= 0
  btile(cw);
  btile(cw + 16);

  for (int j0 = 0; j0 < KL; j0 += 32) {
    // New upper half of the current window [cw+32, cw+64).
    btile(cw + 32);
    btile(cw + 48);

    if (j0 + 32 < KL)
      __builtin_prefetch(khp + (size_t)(j0 + 32) * DHEAD, 0, 1);

    // ---- AC = (q + r_w_bias) . k^T (two 16x16 tiles, K=64 -> 2 WMMAs each)
    v8f s0 = {};
    v8f s1 = {};
    s0 = wmma32(aQW0, wmma_ld_bT(khp + (size_t)j0 * DHEAD, DHEAD), s0);
    s0 = wmma32(aQW1, wmma_ld_bT(khp + (size_t)j0 * DHEAD + 32, DHEAD), s0);
    s1 = wmma32(aQW0, wmma_ld_bT(khp + (size_t)(j0 + 16) * DHEAD, DHEAD), s1);
    s1 = wmma32(aQW1, wmma_ld_bT(khp + (size_t)(j0 + 16) * DHEAD + 32, DHEAD), s1);

    // ---- gather shifted B from the ring, scale (base-2), mask
#pragma unroll
    for (int r = 0; r < 8; ++r) {
      int i = i0 + r + 8 * g;
      int j = j0 + n;
      int c = j + (QL - 1) - i;        // in [cw, cw+62]
      float bd = (c < KL) ? Bb[wid][r + 8 * g][c & 63] : 0.0f;
      float sv = (s0[r] + bd) * SCALE2;
      if (!mp[(size_t)i * KL + j]) sv = NEG_F;
      s0[r] = sv;
      j += 16; c += 16;
      bd = (c < KL) ? Bb[wid][r + 8 * g][c & 63] : 0.0f;
      sv = (s1[r] + bd) * SCALE2;
      if (!mp[(size_t)i * KL + j]) sv = NEG_F;
      s1[r] = sv;
    }

    // ---- online softmax in base 2 (row M lives in one 16-lane half)
#pragma unroll
    for (int r = 0; r < 8; ++r) {
      float mx = fmaxf(s0[r], s1[r]);
      mx = fmaxf(mx, __shfl_xor(mx, 1, 32));
      mx = fmaxf(mx, __shfl_xor(mx, 2, 32));
      mx = fmaxf(mx, __shfl_xor(mx, 4, 32));
      mx = fmaxf(mx, __shfl_xor(mx, 8, 32));
      float mnew = fmaxf(m_r[r], mx);
      float sc = exp2f(m_r[r] - mnew);
      float p0 = exp2f(s0[r] - mnew);
      float p1 = exp2f(s1[r] - mnew);
      float rs = p0 + p1;
      rs += __shfl_xor(rs, 1, 32);
      rs += __shfl_xor(rs, 2, 32);
      rs += __shfl_xor(rs, 4, 32);
      rs += __shfl_xor(rs, 8, 32);
      l_r[r] = l_r[r] * sc + rs;
      m_r[r] = mnew;
#pragma unroll
      for (int t = 0; t < 4; ++t) acc[t][r] *= sc;
      Pt[wid][r + 8 * g][n]      = (_Float16)p0;
      Pt[wid][r + 8 * g][16 + n] = (_Float16)p1;
    }

    // ---- PV: A = P (16x32 from LDS), B = V^T[d-tile, j0..j0+32] contiguous
    v16h ap = wmma_ld_a(&Pt[wid][0][0], 40);
#pragma unroll
    for (int t = 0; t < 4; ++t)
      acc[t] = wmma32(ap, wmma_ld_bT(vhp + (size_t)(t * 16) * KL + j0, KL),
                      acc[t]);

    cw += 32;
  }

  // ---- normalize; emit av in [b, l, h*64+d] f16 layout for the O-gemm
#pragma unroll
  for (int r = 0; r < 8; ++r) {
    float inv = 1.0f / fmaxf(l_r[r], 1e-30f);
    int i = i0 + r + 8 * g;
#pragma unroll
    for (int t = 0; t < 4; ++t)
      av16[((size_t)b * QL + i) * DMODEL + h * DHEAD + t * 16 + n] =
          (_Float16)(acc[t][r] * inv);
  }
}

// ---------------------------------------------------------------------------
extern "C" void kernel_launch(void* const* d_in, const int* in_sizes, int n_in,
                              void* d_out, int out_size, void* d_ws,
                              size_t ws_size, hipStream_t stream) {
  (void)in_sizes; (void)n_in; (void)out_size; (void)ws_size;
  const float* query    = (const float*)d_in[0];
  const float* key      = (const float*)d_in[1];
  const float* value    = (const float*)d_in[2];
  const unsigned char* mask = (const unsigned char*)d_in[3];
  const float* r_emb    = (const float*)d_in[4];
  const float* r_w_bias = (const float*)d_in[5];
  const float* r_bias   = (const float*)d_in[6];
  const float* Wq = (const float*)d_in[7];
  const float* Wk = (const float*)d_in[8];
  const float* Wv = (const float*)d_in[9];
  const float* Wo = (const float*)d_in[10];
  float* out = (float*)d_out;

  // Workspace carve-up (~108 MB total, 256B-aligned slices)
  size_t off = 0;
  auto alloc16 = [&](size_t elems) -> _Float16* {
    _Float16* p = (_Float16*)((char*)d_ws + off);
    off += (elems * sizeof(_Float16) + 255) & ~(size_t)255;
    return p;
  };
  _Float16* q16  = alloc16((size_t)BB * QL * DMODEL);
  _Float16* k16  = alloc16((size_t)BB * KL * DMODEL);
  _Float16* v16_ = alloc16((size_t)BB * KL * DMODEL);
  _Float16* wq16 = alloc16((size_t)DMODEL * DMODEL);
  _Float16* wk16 = alloc16((size_t)DMODEL * DMODEL);
  _Float16* wv16 = alloc16((size_t)DMODEL * DMODEL);
  _Float16* wo16 = alloc16((size_t)DMODEL * DMODEL);
  _Float16* re16 = alloc16((size_t)HH * KL * DHEAD);
  _Float16* qw16 = alloc16((size_t)BB * HH * QL * DHEAD);
  _Float16* qp16 = alloc16((size_t)BB * HH * QL * DHEAD);
  _Float16* kh16 = alloc16((size_t)BB * HH * KL * DHEAD);
  _Float16* vhT16 = alloc16((size_t)BB * HH * DHEAD * KL);
  _Float16* av16 = alloc16((size_t)BB * QL * DMODEL);

  auto cvt = [&](const float* x, _Float16* y, size_t nels) {
    int nt = 256;
    int nb = (int)((nels / 4 + nt - 1) / nt);
    cvt_f16<<<nb, nt, 0, stream>>>(x, y, (int)nels);
  };
  cvt(query, q16,  (size_t)BB * QL * DMODEL);
  cvt(key,   k16,  (size_t)BB * KL * DMODEL);
  cvt(value, v16_, (size_t)BB * KL * DMODEL);
  cvt(Wq, wq16, (size_t)DMODEL * DMODEL);
  cvt(Wk, wk16, (size_t)DMODEL * DMODEL);
  cvt(Wv, wv16, (size_t)DMODEL * DMODEL);
  cvt(Wo, wo16, (size_t)DMODEL * DMODEL);
  cvt(r_emb, re16, (size_t)HH * KL * DHEAD);

  {  // q projection (+bias fold): rows = 4096 -> (rows/32)*(DMODEL/64) tiles
    int rows = BB * QL;
    int tiles = (rows / 32) * (DMODEL / 64);
    gemm_wmma<0><<<tiles / 4, 128, 0, stream>>>(q16, wq16, rows, QL, nullptr,
                                                qw16, qp16, r_w_bias);
  }
  {  // k (head-major) and v (transposed head-major) projections
    int rows = BB * KL;
    int tiles = (rows / 32) * (DMODEL / 64);
    gemm_wmma<1><<<tiles / 4, 128, 0, stream>>>(k16, wk16, rows, KL, nullptr,
                                                kh16, nullptr, nullptr);
    gemm_wmma<3><<<tiles / 4, 128, 0, stream>>>(v16_, wv16, rows, KL, nullptr,
                                                vhT16, nullptr, nullptr);
  }
  {  // flash relative attention
    int waves = BB * HH * (QL / 16);
    attn_wmma<<<waves / 4, 128, 0, stream>>>(qw16, qp16, kh16, vhT16, re16,
                                             r_bias, mask, av16);
  }
  {  // output projection -> f32 result
    int rows = BB * QL;
    int tiles = (rows / 32) * (DMODEL / 64);
    gemm_wmma<2><<<tiles / 4, 128, 0, stream>>>(av16, wo16, rows, QL, out,
                                                nullptr, nullptr, nullptr);
  }
}
